// CoPE_unit_40252433498179
// MI455X (gfx1250) — compile-verified, matrix operand order checked by reading, not run on hardware
//
#include <hip/hip_runtime.h>

// CoPE: out[b,h,s,k] = lerp of (query[b,h,s,:] @ pos_emb) at floor/ceil of
// clamped reversed-cumsum(sigmoid(attn_logits[b,h,s,:])).
//
// MI455X reasoning: 201 MB in + 201 MB out dominates (HBM 23.3 TB/s -> ~19us
// floor). GEMM part is 0.4 GFLOP -> one small WMMA kernel (f16 in, f32 acc).
// Main kernel is pure streaming: float4 loads, wave32 shfl suffix-scan,
// LDS gather, non-temporal float4 stores.

typedef __attribute__((ext_vector_type(16))) _Float16 v16h;
typedef __attribute__((ext_vector_type(8)))  float    v8f;
typedef __attribute__((ext_vector_type(4)))  float    fvec4;

#define D_DIM 64
#define NPOS  64

// ---------------------------------------------------------------------------
// Kernel 1: logits_int[M x 64] = query[M x 64] @ pos_emb[64 x 64]
// One wave32 computes a 16(M) x 64(N) tile via 8x V_WMMA_F32_16X16X32_F16.
// ---------------------------------------------------------------------------
__global__ __launch_bounds__(256) void cope_gemm_wmma(
    const float* __restrict__ query, const float* __restrict__ pos_emb,
    float* __restrict__ logits_int, int M)
{
  const int lane = threadIdx.x & 31;
  const int wave = blockIdx.x * (blockDim.x >> 5) + (threadIdx.x >> 5);
  const int m0   = wave << 4;
  if (m0 >= M) return;                 // wave-uniform: EXEC stays all-1s

  const int msub = lane & 15;
  const int kb   = (lane < 16) ? 0 : 8;   // ISA 16-bit A layout K-base per lane half

  // A: 16x32 f16 per K-chunk. Lane holds row m0+msub, halves K{kb..kb+7} then
  // K{kb+16..kb+23} (per CDNA5 ISA 16-bit A-matrix table).
  const float* q = query + (size_t)(m0 + msub) * D_DIM;
  v16h a[2];
#pragma unroll
  for (int c = 0; c < 2; ++c) {
    const int kc = c * 32;
#pragma unroll
    for (int j = 0; j < 8; ++j) {
      a[c][j]     = (_Float16)q[kc + kb + j];
      a[c][j + 8] = (_Float16)q[kc + kb + 16 + j];
    }
  }

  const v8f zero = {0.f, 0.f, 0.f, 0.f, 0.f, 0.f, 0.f, 0.f};
  v8f acc[4] = {zero, zero, zero, zero};

#pragma unroll
  for (int nb = 0; nb < 4; ++nb) {
    const int n = nb * 16 + msub;      // B: column n in lane (same layout, N for M)
    v16h b0, b1;
#pragma unroll
    for (int j = 0; j < 8; ++j) {
      b0[j]     = (_Float16)pos_emb[(kb + j)      * NPOS + n];
      b0[j + 8] = (_Float16)pos_emb[(kb + 16 + j) * NPOS + n];
      b1[j]     = (_Float16)pos_emb[(32 + kb + j)      * NPOS + n];
      b1[j + 8] = (_Float16)pos_emb[(32 + kb + 16 + j) * NPOS + n];
    }
    acc[nb] = __builtin_amdgcn_wmma_f32_16x16x32_f16(
        false, a[0], false, b0, (short)0, acc[nb], false, false);
    acc[nb] = __builtin_amdgcn_wmma_f32_16x16x32_f16(
        false, a[1], false, b1, (short)0, acc[nb], false, false);
  }

  // D layout: VGPR r -> row r (lanes 0-15) / row r+8 (lanes 16-31), col = lane&15
#pragma unroll
  for (int nb = 0; nb < 4; ++nb) {
    const int n = nb * 16 + msub;
#pragma unroll
    for (int r = 0; r < 8; ++r) {
      const int m = m0 + ((lane < 16) ? r : (r + 8));
      logits_int[(size_t)m * NPOS + n] = acc[nb][r];
    }
  }
}

// ---------------------------------------------------------------------------
// Kernel 2: per row (B*H*S rows of length S=1024): sigmoid -> reversed cumsum
// (suffix scan) -> clamp to 63 -> gather+lerp from this row's 64 logits.
// 256 threads, float4 per thread.  Pure HBM streaming.
// ---------------------------------------------------------------------------
__global__ __launch_bounds__(256) void cope_pos_kernel(
    const float* __restrict__ attn, const float* __restrict__ logits_int,
    float* __restrict__ out)
{
  const int row  = blockIdx.x;
  const int t    = threadIdx.x;
  const int lane = t & 31;
  const int wid  = t >> 5;

  __shared__ float ldsLog[NPOS];
  __shared__ float waveTot[8];

  if (t < NPOS) ldsLog[t] = logits_int[(size_t)row * NPOS + t];

  // Streaming read of one 4KB row (non-temporal: single use).
  const fvec4* arow = (const fvec4*)(attn + (size_t)row * 1024);
  fvec4 g = __builtin_nontemporal_load(arow + t);

  // sigmoid
  float g0 = 1.f / (1.f + __expf(-g[0]));
  float g1 = 1.f / (1.f + __expf(-g[1]));
  float g2 = 1.f / (1.f + __expf(-g[2]));
  float g3 = 1.f / (1.f + __expf(-g[3]));

  // local suffix within the 4 elements (pos[i] = sum_{j>=i} gates[j])
  float s3 = g3;
  float s2 = g2 + s3;
  float s1 = g1 + s2;
  float s0 = g0 + s1;           // = thread total

  // wave32 inclusive suffix scan of thread totals
  float v = s0;
#pragma unroll
  for (int d = 1; d < 32; d <<= 1) {
    float o = __shfl_down(v, d, 32);
    if (lane + d < 32) v += o;
  }
  if (lane == 0) waveTot[wid] = v;   // lane 0 holds whole-wave total
  __syncthreads();                   // also covers ldsLog visibility

  // sum of totals of waves AFTER this one
  float wsuf = 0.f;
#pragma unroll
  for (int w = 0; w < 8; ++w)
    if (w > wid) wsuf += waveTot[w];

  const float excl = (v - s0) + wsuf;  // suffix sum of all later threads

  fvec4 o4;
#pragma unroll
  for (int k = 0; k < 4; ++k) {
    float sk = (k == 0) ? s0 : (k == 1) ? s1 : (k == 2) ? s2 : s3;
    float p  = fminf(sk + excl, 63.0f);      // clamp to NPOS-1
    int   fl = (int)floorf(p);
    int   ce = (int)ceilf(p);
    float w  = p - (float)fl;
    o4[k] = ldsLog[ce] * w + ldsLog[fl] * (1.0f - w);
  }

  fvec4* orow = (fvec4*)(out + (size_t)row * 1024);
  __builtin_nontemporal_store(o4, orow + t);   // write-once output
}

// ---------------------------------------------------------------------------
extern "C" void kernel_launch(void* const* d_in, const int* in_sizes, int n_in,
                              void* d_out, int out_size, void* d_ws, size_t ws_size,
                              hipStream_t stream)
{
  const float* attn    = (const float*)d_in[0];   // [B,H,S,S]
  const float* query   = (const float*)d_in[1];   // [B,H,S,64]
  const float* pos_emb = (const float*)d_in[2];   // [1,64,64]
  float*       out     = (float*)d_out;           // [B,H,S,S]
  float*       logits  = (float*)d_ws;            // M*64 floats = 12 MB scratch

  const int M = in_sizes[1] / D_DIM;              // B*H*S = 49152 rows
  (void)n_in; (void)out_size; (void)ws_size;

  // Kernel 1: 16 rows per wave, 8 waves per block.
  const int waves   = (M + 15) / 16;
  const int blocks1 = (waves + 7) / 8;
  cope_gemm_wmma<<<blocks1, 256, 0, stream>>>(query, pos_emb, logits, M);

  // Kernel 2: one 256-thread block per row of 1024 (float4/thread).
  cope_pos_kernel<<<M, 256, 0, stream>>>(attn, logits, out);
}
